// GIN_4853313044731
// MI455X (gfx1250) — matrix operand order repaired and checked by reading, compiled, not run on hardware
//
#include <hip/hip_runtime.h>
#include <hip/hip_fp16.h>

#define N_    512
#define E_    16384
#define D_    1024
#define L_    3
#define B_    32
#define S_    702
#define ESM_  1280
#define H_    512
#define FLAT_ 492804L
#define MBIG_ (64 * S_)          /* 44928 rows for the fused/seq path */
#define THETA_ 0.08f
#define RES_   0.1f

typedef _Float16 v8h  __attribute__((ext_vector_type(8)));
typedef _Float16 v16h __attribute__((ext_vector_type(16)));
typedef float    v8f  __attribute__((ext_vector_type(8)));
typedef float    v2f  __attribute__((ext_vector_type(2)));
typedef int      v4i_ __attribute__((__vector_size__(16)));   // matches async builtin's param

// ---------- fragment loaders (CDNA5 16-bit WMMA layouts) ----------
// A (16x32, MxK): lane<16 holds row M=lane, K={k..k+7, k+16..k+23};
//                 lane>=16 holds K={k+8..k+15, k+24..k+31}.
__device__ __forceinline__ v16h ld16_gap(const _Float16* p) {
  v8h lo = *(const v8h*)p;
  v8h hi = *(const v8h*)(p + 16);
  return __builtin_shufflevector(lo, hi, 0,1,2,3,4,5,6,7,8,9,10,11,12,13,14,15);
}
// A and B as loaded here (same contiguous k order, not the table order): lane<16 K=k..k+15; lane>=16 K=k+16..k+31.
__device__ __forceinline__ v16h ld16_contig(const _Float16* p) {
  v8h lo = *(const v8h*)p;
  v8h hi = *(const v8h*)(p + 8);
  return __builtin_shufflevector(lo, hi, 0,1,2,3,4,5,6,7,8,9,10,11,12,13,14,15);
}

#define WMMA16(A, Bq, C) \
  __builtin_amdgcn_wmma_f32_16x16x32_f16(false, (A), false, (Bq), (short)0, (C), false, false)

// Async global->LDS staging (gfx1250): guarded so compilation never breaks.
#define ASYNC_OK __has_builtin(__builtin_amdgcn_global_load_async_to_lds_b128)
#if ASYNC_OK
#  if __has_builtin(__builtin_amdgcn_s_wait_asynccnt)
#    define WAIT_ASYNC(n) __builtin_amdgcn_s_wait_asynccnt(n)
#  else
#    define WAIT_ASYNC(n) asm volatile("s_wait_asynccnt %0" :: "i"(n) : "memory")
#  endif
#else
#  define WAIT_ASYNC(n)
#endif

// ---------- generic f16 WMMA GEMM: C = act(scale*(A@B) + bias) ----------
// A: f16 row-major (M x K), optional row-gather: physical row = gather[r/group]*group + r%group
// B: f16 packed column-major [n*K + k]
// 4-wave workgroup computes a 64x128 block tile: waves share the A panel,
// staged in LDS (double-buffered) via GLOBAL_LOAD_ASYNC_TO_LDS_B128 per k-step.
// Out-of-range rows/cols are CLAMPED (not zeroed): they only contaminate C
// elements that are masked at store time.
__global__ void __launch_bounds__(128)
k_gemm_f16(const _Float16* __restrict__ A, long ldA, long strideA,
           const int* __restrict__ gather, int group,
           const _Float16* __restrict__ B, long strideB,
           float* __restrict__ C32, _Float16* __restrict__ C16, long strideC,
           const float* __restrict__ bias,
           int M, int Nn, int K, int act, float scale)
{
  __shared__ _Float16 sA[2][64 * 32];   // 2 x 4 KB: 64 rows x 32 halves (one k-step)

  const int tid  = threadIdx.x;
  const int lane = tid & 31;
  const int wv   = tid >> 5;            // wave 0..3 -> N sub-tile
  const int l15  = lane & 15;
  const int hsel = lane >> 4;
  const int m0 = blockIdx.y * 64;
  const int n0 = blockIdx.x * 128 + wv * 32;
  A += (long)blockIdx.z * strideA;
  B += (long)blockIdx.z * strideB;
  const long cOff = (long)blockIdx.z * strideC;

  // ---- staging: 64 rows x 64B = 256 16B segments, 2 per thread ----
  const int srow0 = tid >> 2;           // rows 0..31
  const int srow1 = srow0 + 32;         // rows 32..63
  const int spart = (tid & 3) * 8;      // half-offset within a 32-half row
  auto arow = [&](int rr) -> const _Float16* {
    int r = m0 + rr;
    if (r > M - 1) r = M - 1;
    long row = gather ? ((long)gather[r / group] * (long)group + (long)(r % group))
                      : (long)r;
    return A + row * ldA + spart;
  };
  const _Float16* ga0 = arow(srow0);
  const _Float16* ga1 = arow(srow1);
  _Float16* sd0[2] = { &sA[0][srow0 * 32 + spart], &sA[1][srow0 * 32 + spart] };
  _Float16* sd1[2] = { &sA[0][srow1 * 32 + spart], &sA[1][srow1 * 32 + spart] };

  auto stage = [&](int buf, int k) {
#if ASYNC_OK
    __builtin_amdgcn_global_load_async_to_lds_b128(
        (v4i_*)(ga0 + k),
        (__attribute__((address_space(3))) v4i_*)sd0[buf], 0, 0);
    __builtin_amdgcn_global_load_async_to_lds_b128(
        (v4i_*)(ga1 + k),
        (__attribute__((address_space(3))) v4i_*)sd1[buf], 0, 0);
#else
    *(v8h*)sd0[buf] = *(const v8h*)(ga0 + k);
    *(v8h*)sd1[buf] = *(const v8h*)(ga1 + k);
#endif
  };

  // ---- B per-lane pointers (direct global, served by 192MB L2) ----
  auto bcol = [&](int c) -> const _Float16* {
    if (c > Nn - 1) c = Nn - 1;
    return B + (long)c * K + (hsel << 4);
  };
  const _Float16* bp0 = bcol(n0 + l15);
  const _Float16* bp1 = bcol(n0 + 16 + l15);

  // ---- A fragment offsets inside the staged panel ----
  const int fa0 = (l15     ) * 32 + hsel * 16;
  const int fa1 = (l15 + 16) * 32 + hsel * 16;
  const int fa2 = (l15 + 32) * 32 + hsel * 16;
  const int fa3 = (l15 + 48) * 32 + hsel * 16;

  v8f acc00 = {}, acc01 = {}, acc10 = {}, acc11 = {};
  v8f acc20 = {}, acc21 = {}, acc30 = {}, acc31 = {};

  stage(0, 0);
  int buf = 0;
  for (int k = 0; k < K; k += 32) {
    const bool more = (k + 32) < K;
    if (more) stage(buf ^ 1, k + 32);
    if (more) { WAIT_ASYNC(2); } else { WAIT_ASYNC(0); }
    __syncthreads();                     // panel visible to all 4 waves
    const _Float16* sb = sA[buf];
    v16h a0 = ld16_contig(sb + fa0);
    v16h a1 = ld16_contig(sb + fa1);
    v16h a2 = ld16_contig(sb + fa2);
    v16h a3 = ld16_contig(sb + fa3);
    v16h b0 = ld16_contig(bp0 + k);
    v16h b1 = ld16_contig(bp1 + k);
    acc00 = WMMA16(a0, b0, acc00);  acc01 = WMMA16(a0, b1, acc01);
    acc10 = WMMA16(a1, b0, acc10);  acc11 = WMMA16(a1, b1, acc11);
    acc20 = WMMA16(a2, b0, acc20);  acc21 = WMMA16(a2, b1, acc21);
    acc30 = WMMA16(a3, b0, acc30);  acc31 = WMMA16(a3, b1, acc31);
    __syncthreads();                     // readers done before buffer reuse
    buf ^= 1;
  }

  // C/D layout: lane<16 -> (M=r, N=lane); lane>=16 -> (M=8+r, N=lane-16)
#define STORE_TILE(ACC, MT, NT) do {                                          \
    int n = n0 + (NT) * 16 + l15;                                             \
    if (n < Nn) {                                                             \
      int mb = m0 + (MT) * 16 + (hsel << 3);                                  \
      float bv = bias ? bias[n] : 0.f;                                        \
      _Pragma("unroll")                                                       \
      for (int rr = 0; rr < 8; rr++) {                                        \
        int m = mb + rr;                                                      \
        if (m < M) {                                                          \
          float v = (ACC)[rr] * scale + bv;                                   \
          if (act == 1)      v = fmaxf(v, 0.f);                               \
          else if (act == 2) v = 1.f / (1.f + __expf(-v));                    \
          long o = cOff + (long)m * Nn + n;                                   \
          if (C32) C32[o] = v;                                                \
          if (C16) C16[o] = (_Float16)v;                                      \
        }                                                                     \
      }                                                                       \
    }                                                                         \
  } while (0)

  STORE_TILE(acc00, 0, 0);  STORE_TILE(acc01, 0, 1);
  STORE_TILE(acc10, 1, 0);  STORE_TILE(acc11, 1, 1);
  STORE_TILE(acc20, 2, 0);  STORE_TILE(acc21, 2, 1);
  STORE_TILE(acc30, 3, 0);  STORE_TILE(acc31, 3, 1);
#undef STORE_TILE
}

// ---------- c1: (32 x FLAT) @ (FLAT x 512), fp32 WMMA 16x16x4, split-K + atomics ----------
#define C1_SPLITS 128
#define C1_KC     3852
__global__ void k_c1(const float* __restrict__ F, const float* __restrict__ W,
                     float* __restrict__ Cacc)
{
  const int lane = threadIdx.x;
  const int l15 = lane & 15, hsel = lane >> 4;
  const int c = blockIdx.x * 16 + l15;          // N column (<512)
  long k0 = (long)blockIdx.y * C1_KC;
  long kend = k0 + C1_KC; if (kend > FLAT_) kend = FLAT_;
  // running pointers (A: lane<16 K={k,k+1}; lane>=16 K={k+2,k+3})
  const float* f0 = F + (long)l15 * FLAT_ + k0 + hsel * 2;
  const float* f1 = F + (long)(l15 + 16) * FLAT_ + k0 + hsel * 2;
  const float* wb = W + ((long)k0 + hsel * 2) * 512 + c;
  v8f acc0 = {}, acc1 = {};
  long steps = (kend - k0) >> 2;
  #pragma unroll 2
  for (long s = 0; s < steps; s++) {
    v2f a0 = *(const v2f*)f0;
    v2f a1 = *(const v2f*)f1;
    v2f b; b[0] = wb[0]; b[1] = wb[512];
    acc0 = __builtin_amdgcn_wmma_f32_16x16x4_f32(false, a0, false, b, (short)0, acc0, false, false);
    acc1 = __builtin_amdgcn_wmma_f32_16x16x4_f32(false, a1, false, b, (short)0, acc1, false, false);
    f0 += 4; f1 += 4; wb += 4 * 512;
  }
  const int mb = hsel * 8;
  #pragma unroll
  for (int r = 0; r < 8; r++) {
    atomicAdd(&Cacc[(mb + r) * 512 + c],      acc0[r]);
    atomicAdd(&Cacc[(16 + mb + r) * 512 + c], acc1[r]);
  }
}

// ---------- small helpers ----------
__global__ void k_pack_w(const float* __restrict__ W, _Float16* __restrict__ out,
                         int K, int Nn, long total) {
  long i = (long)blockIdx.x * 256 + threadIdx.x;
  if (i >= total) return;
  long n = i / K, k = i % K;
  out[i] = (_Float16)W[k * (long)Nn + n];      // column-major f16 pack
}
__global__ void k_zero(float* p, long total) {
  long i = (long)blockIdx.x * 256 + threadIdx.x;
  if (i < total) p[i] = 0.f;
}
__global__ void k_init_h(const float* __restrict__ x, float* __restrict__ h,
                         _Float16* __restrict__ xh, long total) {
  long i = (long)blockIdx.x * 256 + threadIdx.x;
  if (i >= total) return;
  h[i] = x[i]; xh[i] = (_Float16)x[i];
}
__global__ void k_edge_agg(const float* __restrict__ h, const int* __restrict__ src,
                           const int* __restrict__ dst, float* __restrict__ agg) {
  int e = blockIdx.x;
  int d = threadIdx.x * 4;
  const float4 v = *(const float4*)(h + (long)src[e] * D_ + d);
  float* a = agg + (long)dst[e] * D_ + d;
  atomicAdd(a + 0, v.x); atomicAdd(a + 1, v.y);
  atomicAdd(a + 2, v.z); atomicAdd(a + 3, v.w);
}
__global__ void k_z0(const float* __restrict__ h, const float* __restrict__ agg,
                     const float* __restrict__ eps, int l, _Float16* __restrict__ z, long total) {
  long i = (long)blockIdx.x * 256 + threadIdx.x;
  if (i >= total) return;
  z[i] = (_Float16)((1.f + eps[l]) * h[i] + agg[i]);
}
// bn affine + theta*xi (+ res*h) then LayerNorm(D=1024); writes f32 + f16
__global__ void k_post(const float* __restrict__ z2, const float* __restrict__ xi,
                       float* __restrict__ h32, _Float16* __restrict__ h16,
                       const float* __restrict__ bng, const float* __restrict__ bnb,
                       const float* __restrict__ bnm, const float* __restrict__ bnv,
                       const float* __restrict__ lng, const float* __restrict__ lnb,
                       int has_res) {
  int row = blockIdx.x, tid = threadIdx.x;
  __shared__ float red[256];
  long base = (long)row * D_;
  float y[4];
  #pragma unroll
  for (int j = 0; j < 4; j++) {
    int i = tid + j * 256;
    float v = z2[base + i];
    v = bng[i] * (v - bnm[i]) * rsqrtf(bnv[i] + 1e-5f) + bnb[i];
    v += THETA_ * xi[base + i];
    if (has_res) v += RES_ * h32[base + i];
    y[j] = v;
  }
  red[tid] = y[0] + y[1] + y[2] + y[3]; __syncthreads();
  for (int o = 128; o > 0; o >>= 1) { if (tid < o) red[tid] += red[tid + o]; __syncthreads(); }
  float mean = red[0] * (1.f / D_); __syncthreads();
  float s2 = 0;
  #pragma unroll
  for (int j = 0; j < 4; j++) { float d = y[j] - mean; s2 += d * d; }
  red[tid] = s2; __syncthreads();
  for (int o = 128; o > 0; o >>= 1) { if (tid < o) red[tid] += red[tid + o]; __syncthreads(); }
  float inv = rsqrtf(red[0] * (1.f / D_) + 1e-5f);
  #pragma unroll
  for (int j = 0; j < 4; j++) {
    int i = tid + j * 256;
    float o = lng[i] * (y[j] - mean) * inv + lnb[i];
    h32[base + i] = o; h16[base + i] = (_Float16)o;
  }
}
__global__ void k_nodes(const int* __restrict__ ppi, const int* __restrict__ idx,
                        int* __restrict__ nodes) {
  int b = threadIdx.x;
  if (b >= 64) return;
  int col = (b < 32) ? 0 : 1;
  int bb  = (b < 32) ? b : b - 32;
  nodes[b] = ppi[idx[bb] * 2 + col];
}
__global__ void k_fused(const float* __restrict__ gate, const float* __restrict__ sp,
                        const float* __restrict__ sq, _Float16* __restrict__ out, long total) {
  long i = (long)blockIdx.x * 256 + threadIdx.x;
  if (i >= total) return;
  int d = (int)(i & 511);
  int b = (int)((i >> 9) / S_);
  float g = gate[b * H_ + d];
  out[i] = (_Float16)(g * sp[b * H_ + d] + (1.f - g) * sq[i]);
}
// LayerNorm(H=512) then L2-normalize; input T already relu'd by GEMM epilogue
__global__ void k_lnl2n(const float* __restrict__ T, const float* __restrict__ g,
                        const float* __restrict__ b, _Float16* __restrict__ out) {
  int row = blockIdx.x, tid = threadIdx.x;
  __shared__ float red[256];
  long base = (long)row * H_;
  float y0 = T[base + tid], y1 = T[base + tid + 256];
  red[tid] = y0 + y1; __syncthreads();
  for (int o = 128; o > 0; o >>= 1) { if (tid < o) red[tid] += red[tid + o]; __syncthreads(); }
  float mean = red[0] * (1.f / H_); __syncthreads();
  float d0 = y0 - mean, d1 = y1 - mean;
  red[tid] = d0 * d0 + d1 * d1; __syncthreads();
  for (int o = 128; o > 0; o >>= 1) { if (tid < o) red[tid] += red[tid + o]; __syncthreads(); }
  float inv = rsqrtf(red[0] * (1.f / H_) + 1e-5f); __syncthreads();
  float o0 = g[tid] * d0 * inv + b[tid];
  float o1 = g[tid + 256] * d1 * inv + b[tid + 256];
  red[tid] = o0 * o0 + o1 * o1; __syncthreads();
  for (int o = 128; o > 0; o >>= 1) { if (tid < o) red[tid] += red[tid + o]; __syncthreads(); }
  float s = 1.f / fmaxf(sqrtf(red[0]), 1e-12f);
  out[base + tid]       = (_Float16)(o0 * s);
  out[base + tid + 256] = (_Float16)(o1 * s);
}
__global__ void k_out1(const float* __restrict__ C, const float* __restrict__ bias,
                       float* __restrict__ out1) {
  int i = blockIdx.x * 256 + threadIdx.x;
  if (i < 32 * H_) out1[i] = fmaxf(C[i] + bias[i & 511], 0.f);
}
__global__ void k_c2(const float* __restrict__ a, const float* __restrict__ w,
                     const float* __restrict__ b, float* __restrict__ out2) {
  int m = blockIdx.x, n = threadIdx.x;
  float acc = b[n];
  for (int k = 0; k < H_; k++) acc += a[m * H_ + k] * w[k * 256 + n];
  out2[m * 256 + n] = fmaxf(acc, 0.f);
}
__global__ void k_c3(const float* __restrict__ a, const float* __restrict__ w,
                     const float* __restrict__ b, float* __restrict__ out) {
  int m = threadIdx.x;
  if (m >= 32) return;
  float acc = b[0];
  for (int k = 0; k < 256; k++) acc += a[m * 256 + k] * w[k];
  out[m] = acc;
}

// ------------------------------- host -------------------------------
extern "C" void kernel_launch(void* const* d_in, const int* in_sizes, int n_in,
                              void* d_out, int out_size, void* d_ws, size_t ws_size,
                              hipStream_t stream)
{
  (void)in_sizes; (void)n_in; (void)out_size; (void)ws_size;
  const float*    x        = (const float*)d_in[0];
  const int*      edge_src = (const int*)d_in[1];
  const int*      edge_dst = (const int*)d_in[2];
  const _Float16* esm      = (const _Float16*)d_in[3];
  const int*      ppi      = (const int*)d_in[4];
  const int*      idx      = (const int*)d_in[5];
  const float* ip_w=(const float*)d_in[6],  *ip_b=(const float*)d_in[7];
  const float* eps =(const float*)d_in[8];
  const float* w1  =(const float*)d_in[9],  *b1  =(const float*)d_in[10];
  const float* w2  =(const float*)d_in[11], *b2  =(const float*)d_in[12];
  const float* bn_g=(const float*)d_in[13], *bn_b=(const float*)d_in[14];
  const float* bn_m=(const float*)d_in[15], *bn_v=(const float*)d_in[16];
  const float* ln_g=(const float*)d_in[17], *ln_b=(const float*)d_in[18];
  const float* lin_w=(const float*)d_in[19],*lin_b=(const float*)d_in[20];
  const float* sp_w=(const float*)d_in[21], *sp_b=(const float*)d_in[22];
  const float* sq_w=(const float*)d_in[23], *sq_b=(const float*)d_in[24];
  const float* g1_w=(const float*)d_in[25], *g1_b=(const float*)d_in[26];
  const float* g2_w=(const float*)d_in[27], *g2_b=(const float*)d_in[28];
  const float* fo_w=(const float*)d_in[29], *fo_b=(const float*)d_in[30];
  const float* fo_g=(const float*)d_in[31], *fo_bb=(const float*)d_in[32];
  const float* c1_w=(const float*)d_in[33], *c1_b=(const float*)d_in[34];
  const float* c2_w=(const float*)d_in[35], *c2_b=(const float*)d_in[36];
  const float* c3_w=(const float*)d_in[37], *c3_b=(const float*)d_in[38];

  char* base = (char*)d_ws;
  size_t off = 0;
  auto carve = [&](size_t bytes) -> void* {
    size_t o = off; off += (bytes + 255) & ~(size_t)255; return base + o;
  };
  _Float16* wp_ip  = (_Float16*)carve((size_t)D_*D_*2);
  _Float16* wp_w1  = (_Float16*)carve((size_t)L_*D_*D_*2);
  _Float16* wp_w2  = (_Float16*)carve((size_t)L_*D_*D_*2);
  _Float16* wp_lin = (_Float16*)carve((size_t)D_*D_*2);
  _Float16* wp_sp  = (_Float16*)carve((size_t)D_*H_*2);
  _Float16* wp_sq  = (_Float16*)carve((size_t)ESM_*H_*2);
  _Float16* wp_g1  = (_Float16*)carve((size_t)H_*H_*2);
  _Float16* wp_g2  = (_Float16*)carve((size_t)H_*H_*2);
  _Float16* wp_fo  = (_Float16*)carve((size_t)H_*H_*2);
  _Float16* xh     = (_Float16*)carve((size_t)N_*D_*2);
  float*    xi     = (float*)carve((size_t)N_*D_*4);
  float*    h32    = (float*)carve((size_t)N_*D_*4);
  float*    agg    = (float*)carve((size_t)N_*D_*4);
  float*    z2     = (float*)carve((size_t)N_*D_*4);
  _Float16* z0h    = (_Float16*)carve((size_t)N_*D_*2);
  _Float16* z1h    = (_Float16*)carve((size_t)N_*D_*2);
  _Float16* h16    = (_Float16*)carve((size_t)N_*D_*2);
  _Float16* hlin16 = (_Float16*)carve((size_t)N_*D_*2);
  int*      nodes  = (int*)carve(64*4);
  float*    sp32   = (float*)carve((size_t)64*H_*4);
  float*    gate32 = (float*)carve((size_t)64*H_*4);
  _Float16* sp16   = (_Float16*)carve((size_t)64*H_*2);
  _Float16* t16    = (_Float16*)carve((size_t)64*H_*2);
  float*    big32  = (float*)carve((size_t)MBIG_*H_*4);   // sq out, then fo out
  _Float16* big16  = (_Float16*)carve((size_t)MBIG_*H_*2);// fused, then a (normalized)
  float*    c1acc  = (float*)carve((size_t)32*H_*4);
  float*    out1   = (float*)carve((size_t)32*H_*4);
  float*    out2   = (float*)carve((size_t)32*256*4);

  auto cdiv = [](long a, long b) { return (unsigned)((a + b - 1) / b); };
  auto gemm = [&](const _Float16* Aq, long ldA, long sA, const int* gth, int grp,
                  const _Float16* Bq, long sB, float* C32, _Float16* C16, long sC,
                  const float* bias, int M, int Nn, int K, int act, float scale, int batch) {
    dim3 g(cdiv(Nn, 128), cdiv(M, 64), (unsigned)batch);
    k_gemm_f16<<<g, 128, 0, stream>>>(Aq, ldA, sA, gth, grp, Bq, sB,
                                      C32, C16, sC, bias, M, Nn, K, act, scale);
  };

  // ---- pack weights to f16 column-major ----
  k_pack_w<<<cdiv((long)D_*D_,256),256,0,stream>>>(ip_w,  wp_ip,  D_, D_, (long)D_*D_);
  for (int l = 0; l < L_; l++) {
    k_pack_w<<<cdiv((long)D_*D_,256),256,0,stream>>>(w1 + (long)l*D_*D_, wp_w1 + (long)l*D_*D_, D_, D_, (long)D_*D_);
    k_pack_w<<<cdiv((long)D_*D_,256),256,0,stream>>>(w2 + (long)l*D_*D_, wp_w2 + (long)l*D_*D_, D_, D_, (long)D_*D_);
  }
  k_pack_w<<<cdiv((long)D_*D_,256),256,0,stream>>>(lin_w, wp_lin, D_, D_, (long)D_*D_);
  k_pack_w<<<cdiv((long)D_*H_,256),256,0,stream>>>(sp_w,  wp_sp,  D_, H_, (long)D_*H_);
  k_pack_w<<<cdiv((long)ESM_*H_,256),256,0,stream>>>(sq_w, wp_sq, ESM_, H_, (long)ESM_*H_);
  k_pack_w<<<cdiv((long)H_*H_,256),256,0,stream>>>(g1_w,  wp_g1,  H_, H_, (long)H_*H_);
  k_pack_w<<<cdiv((long)H_*H_,256),256,0,stream>>>(g2_w,  wp_g2,  H_, H_, (long)H_*H_);
  k_pack_w<<<cdiv((long)H_*H_,256),256,0,stream>>>(fo_w,  wp_fo,  H_, H_, (long)H_*H_);

  // ---- GIN stack ----
  const long ND = (long)N_ * D_;
  k_init_h<<<cdiv(ND,256),256,0,stream>>>(x, h32, xh, ND);
  gemm(xh, D_, 0, nullptr, 1, wp_ip, 0, xi, nullptr, 0, ip_b, N_, D_, D_, 0, 1.f, 1);
  for (int l = 0; l < L_; l++) {
    k_zero<<<cdiv(ND,256),256,0,stream>>>(agg, ND);
    k_edge_agg<<<E_,256,0,stream>>>(h32, edge_src, edge_dst, agg);
    k_z0<<<cdiv(ND,256),256,0,stream>>>(h32, agg, eps, l, z0h, ND);
    gemm(z0h, D_, 0, nullptr, 1, wp_w1 + (long)l*D_*D_, 0, nullptr, z1h, 0, b1 + l*D_, N_, D_, D_, 1, 1.f, 1);
    gemm(z1h, D_, 0, nullptr, 1, wp_w2 + (long)l*D_*D_, 0, z2, nullptr, 0, b2 + l*D_, N_, D_, D_, 1, 1.f, 1);
    k_post<<<N_,256,0,stream>>>(z2, xi, h32, h16,
                                bn_g + l*D_, bn_b + l*D_, bn_m + l*D_, bn_v + l*D_,
                                ln_g + l*D_, ln_b + l*D_, l > 0 ? 1 : 0);
  }
  gemm(h16, D_, 0, nullptr, 1, wp_lin, 0, nullptr, hlin16, 0, lin_b, N_, D_, D_, 1, 1.f, 1);

  // ---- pair gather + fusion path (x1 rows 0..31, x2 rows 32..63) ----
  k_nodes<<<1,64,0,stream>>>(ppi, idx, nodes);
  gemm(hlin16, D_, 0, nodes, 1,   wp_sp, 0, sp32, sp16, 0, sp_b, 64, H_, D_, 1, 1.f, 1);
  gemm(sp16,   H_, 0, nullptr, 1, wp_g1, 0, nullptr, t16, 0, g1_b, 64, H_, H_, 1, 1.f, 1);
  gemm(t16,    H_, 0, nullptr, 1, wp_g2, 0, gate32, nullptr, 0, g2_b, 64, H_, H_, 2, 1.f, 1);
  gemm(esm, ESM_, 0, nodes, S_,   wp_sq, 0, big32, nullptr, 0, sq_b, MBIG_, H_, ESM_, 1, 1.f, 1);
  const long BIG = (long)MBIG_ * H_;
  k_fused<<<cdiv(BIG,256),256,0,stream>>>(gate32, sp32, big32, big16, BIG);
  gemm(big16, H_, 0, nullptr, 1, wp_fo, 0, big32, nullptr, 0, fo_b, MBIG_, H_, H_, 1, 1.f, 1);
  k_lnl2n<<<MBIG_,256,0,stream>>>(big32, fo_g, fo_bb, big16);   // big16 now = normalized a

  // ---- inter = batched a1 @ a2^T / sqrt(H) -> d_out[32..] ----
  float* interp = ((float*)d_out) + 32;
  gemm(big16, H_, (long)S_*H_, nullptr, 1,
       big16 + (long)32*S_*H_, (long)S_*H_,
       interp, nullptr, (long)S_*S_,
       nullptr, S_, S_, H_, 0, 0.044194173824159216f /* 1/sqrt(512) */, 32);

  // ---- classifier head: c1 (f32 WMMA split-K), c2, c3 ----
  k_zero<<<cdiv(32L*H_,256),256,0,stream>>>(c1acc, 32L*H_);
  k_c1<<<dim3(H_/16, C1_SPLITS), 32, 0, stream>>>(interp, c1_w, c1acc);
  k_out1<<<cdiv(32L*H_,256),256,0,stream>>>(c1acc, c1_b, out1);
  k_c2<<<32,256,0,stream>>>(out1, c2_w, c2_b, out2);
  k_c3<<<1,32,0,stream>>>(out2, c3_w, c3_b, (float*)d_out);
}
